// SwinTransformerBlock_ADALN_3865470566407
// MI455X (gfx1250) — matrix operand order repaired and checked by reading, not compiled, not run on hardware
//
#include <hip/hip_runtime.h>

// ---------------------------------------------------------------------------
// Types / helpers
// ---------------------------------------------------------------------------
typedef __attribute__((ext_vector_type(16))) __bf16 v16bf;
typedef __attribute__((ext_vector_type(8)))  float  v8f;
typedef __attribute__((ext_vector_type(4)))  unsigned int u32x4;
typedef __attribute__((ext_vector_type(8)))  int i32x8;
typedef __attribute__((ext_vector_type(4)))  int i32x4;

struct __align__(16) U4 { unsigned int x, y, z, w; };
union FragBF { v16bf v; U4 q[2]; unsigned short s[16]; };

#if defined(__has_builtin)
#if __has_builtin(__builtin_amdgcn_tensor_load_to_lds) && \
    __has_builtin(__builtin_amdgcn_s_wait_tensorcnt)
#define HAVE_TDM 1
#endif
#endif
#ifndef HAVE_TDM
#define HAVE_TDM 0
#endif

__device__ __forceinline__ unsigned short f2bf(float f) {
  unsigned int u = __float_as_uint(f);
  u += 0x7fffu + ((u >> 16) & 1u);          // round-to-nearest-even
  return (unsigned short)(u >> 16);
}

__device__ __forceinline__ v8f wmma_bf16(v16bf a, v16bf b, v8f c) {
  return __builtin_amdgcn_wmma_f32_16x16x32_bf16(false, a, false, b,
                                                 (short)0, c, false, false);
}

// Two contiguous 16B chunks -> one 16x32 A / 32x16 B fragment per-lane half.
__device__ __forceinline__ v16bf load_frag(const unsigned short* p0,
                                           const unsigned short* p1) {
  FragBF f;
  f.q[0] = *(const U4*)p0;
  f.q[1] = *(const U4*)p1;
  return f.v;
}

// Low 32 bits of an LDS flat address are the LDS byte offset (ISA 10.2).
__device__ __forceinline__ unsigned lds_off(const void* p) {
  return (unsigned)(unsigned long long)(size_t)p;
}

#if HAVE_TDM
// TDM: DMA a 2-D tile (rows x 32 bf16 elements, row stride K elements) from
// global into LDS.  D# bitfields per CDNA5 ISA sec. 8 (groups 0/1; 2-D tensor
// so groups 2/3 are zero).  Tracked by TENSORcnt.
__device__ __forceinline__ void tdm_load_b_tile(const unsigned short* gsrc,
                                                unsigned ldsaddr, int K) {
  unsigned long long ga = (unsigned long long)(size_t)gsrc;
  u32x4 g0;
  g0.x = 1u;                                         // count=1, user mode
  g0.y = ldsaddr;                                    // lds_addr (bytes)
  g0.z = (unsigned)ga;                               // global_addr[31:0]
  g0.w = (unsigned)((ga >> 32) & 0x01ffffffu)        // global_addr[56:32]
         | 0x80000000u;                              // type=2 ("image")
  i32x8 g1;
  g1[0] = 0x00010000;                                // data_size=1 (2 bytes)
  g1[1] = (int)(((unsigned)K & 0xffffu) << 16);      // tensor_dim0 lo16
  g1[2] = (int)((((unsigned)K >> 16) & 0xffffu)      // tensor_dim0 hi16
                | (64u << 16));                      // tensor_dim1 = 64
  g1[3] = (int)(32u << 16);                          // tile_dim0 = 32
  g1[4] = 64;                                        // tile_dim1 = 64
  g1[5] = K;                                         // dim0_stride lo32
  g1[6] = 0;                                         // dim0_stride hi16
  g1[7] = 0;
  i32x4 z4 = {0, 0, 0, 0};
#if __clang_major__ >= 23
  i32x8 z8 = {0, 0, 0, 0, 0, 0, 0, 0};
  __builtin_amdgcn_tensor_load_to_lds(g0, g1, z4, z4, z8, 0);
#else
  __builtin_amdgcn_tensor_load_to_lds(g0, g1, z4, z4, 0);
#endif
}
#endif

// Stage one 64x32 bf16 B-tile into LDS: TDM (wave 0 issues, DMA engine copies)
// or cooperative fallback.
__device__ __forceinline__ void stage_b_tile(const unsigned short* Wtile, int K,
                                             unsigned short (*Bs)[32], int tid) {
#if HAVE_TDM
  if (tid < 32) tdm_load_b_tile(Wtile, lds_off(&Bs[0][0]), K);
#else
  int row = tid >> 2, part = tid & 3;
  *(U4*)&Bs[row][part * 8] = *(const U4*)(Wtile + (size_t)row * K + part * 8);
#endif
}

__device__ __forceinline__ void stage_b_wait(int tid) {
#if HAVE_TDM
  if (tid < 32) __builtin_amdgcn_s_wait_tensorcnt(0);
#else
  (void)tid;  // ds stores drained by the barrier's implicit waits
#endif
}

// ---------------------------------------------------------------------------
// adaLN: ada[b, 0:3072] = silu(cond[b]) @ ada_w^T + ada_b
// grid 384 (= 32 b * 12 chunks), 256 thr; silu(cond) staged in LDS.
// ---------------------------------------------------------------------------
__global__ __launch_bounds__(256) void ada_kernel(
    const float* __restrict__ cond, const float* __restrict__ aw,
    const float* __restrict__ ab, float* __restrict__ ada) {
  __shared__ float sc[512];
  int b = blockIdx.x / 12;
  int j = (blockIdx.x % 12) * 256 + threadIdx.x;
  for (int i = threadIdx.x; i < 512; i += 256) {
    float c = cond[b * 512 + i];
    sc[i] = c / (1.f + __expf(-c));
  }
  __syncthreads();
  const float* wr = aw + (size_t)j * 512;
  float sum = 0.f;
#pragma unroll 8
  for (int k = 0; k < 512; ++k) sum += sc[k] * wr[k];
  ada[(size_t)b * 3072 + j] = sum + ab[j];
}

// fp32 -> bf16 weight conversion
__global__ __launch_bounds__(256) void cvt_kernel(
    const float* __restrict__ in, unsigned short* __restrict__ out, int n) {
  int i = blockIdx.x * 256 + threadIdx.x;
  if (i < n) out[i] = f2bf(in[i]);
}

// ---------------------------------------------------------------------------
// LN + adaLN modulate (+ cyclic shift & window partition for PHASE 1).
// One wave per destination token; 16 channels/lane; wave32 shuffle reduce.
// ---------------------------------------------------------------------------
template <int PHASE>
__global__ __launch_bounds__(256) void prep_kernel(
    const float* __restrict__ src, const float* __restrict__ ada,
    const float* __restrict__ gw, const float* __restrict__ gb,
    unsigned short* __restrict__ dst) {
  int wave = threadIdx.x >> 5, lane = threadIdx.x & 31;
  int tok = blockIdx.x * 8 + wave;
  int b;
  size_t srow;
  if (PHASE == 1) {
    int w = tok >> 6, t = tok & 63;
    b = w >> 4;
    int wi = w & 15;
    int hp = ((wi >> 2) << 3) + (t >> 3);      // rolled row
    int wp = ((wi & 3) << 3) + (t & 7);        // rolled col
    int h = (hp + 4) & 31, ww = (wp + 4) & 31; // undo roll(-4) to find source
    srow = ((size_t)b * 1024 + h * 32 + ww) * 512;
  } else {
    b = tok >> 10;
    srow = (size_t)tok * 512;
  }
  const float4* px = (const float4*)(src + srow);
  float vals[16];
  float s = 0.f, sq = 0.f;
#pragma unroll
  for (int j = 0; j < 4; ++j) {
    float4 v = px[lane * 4 + j];
    vals[4 * j + 0] = v.x; vals[4 * j + 1] = v.y;
    vals[4 * j + 2] = v.z; vals[4 * j + 3] = v.w;
    s += v.x + v.y + v.z + v.w;
    sq += v.x * v.x + v.y * v.y + v.z * v.z + v.w * v.w;
  }
#pragma unroll
  for (int o = 16; o > 0; o >>= 1) {
    s += __shfl_xor(s, o);
    sq += __shfl_xor(sq, o);
  }
  float mu = s * (1.f / 512.f);
  float rstd = rsqrtf(sq * (1.f / 512.f) - mu * mu + 1e-5f);
  const float* sh = ada + (size_t)b * 3072 + (PHASE == 1 ? 0 : 1536);
  const float* scm = sh + 512;
  int c0 = lane * 16;
  unsigned int pk[8];
#pragma unroll
  for (int j = 0; j < 16; ++j) {
    int c = c0 + j;
    float y = (vals[j] - mu) * rstd * gw[c] + gb[c];
    y = y * (1.f + scm[c]) + sh[c];
    unsigned short h16 = f2bf(y);
    if (j & 1) pk[j >> 1] |= ((unsigned int)h16) << 16;
    else       pk[j >> 1] = h16;
  }
  U4* pd = (U4*)(dst + (size_t)tok * 512 + c0);
  U4 t0 = {pk[0], pk[1], pk[2], pk[3]};
  U4 t1 = {pk[4], pk[5], pk[6], pk[7]};
  pd[0] = t0;
  pd[1] = t1;
}

// ---------------------------------------------------------------------------
// WMMA GEMM:  D[m,n] = sum_k A[m,k] * W[n,k] + bias[n]  (A bf16 MxK, W bf16 NxK)
// 256 thr = 8 waves; block tile 256(M) x 64(N); wave tile 32 x 64; K step 32.
// Shared 64x32 B-tile staged in LDS by the Tensor Data Mover, double-buffered
// so the DMA overlaps the WMMA stream.  8 WMMAs per wave per K-step.
// MODE 0: store bf16                                               (qkv)
// MODE 1: window-reverse + roll scatter, x2 = x + g_msa*val, fp32  (proj)
// MODE 2: exact GELU, store bf16                                   (fc1)
// MODE 3: out = x2 + g_mlp*val, fp32                               (fc2)
// ---------------------------------------------------------------------------
template <int MODE>
__global__ __launch_bounds__(256) void gemm_bf16_kernel(
    const unsigned short* __restrict__ A, const unsigned short* __restrict__ W,
    const float* __restrict__ bias, unsigned short* __restrict__ outb,
    const float* __restrict__ ada, const float* __restrict__ xin,
    float* __restrict__ outf, int M, int N, int K) {
  __shared__ __align__(16) unsigned short Bs[2][64][32];
  int tid = threadIdx.x;
  int wave = tid >> 5, lane = tid & 31;
  int laneHi = lane >> 4, ln = lane & 15;
  int m0 = blockIdx.y * 256 + wave * 32;
  int n0 = blockIdx.x * 64;
  const unsigned short* arow0 = A + (size_t)(m0 + ln) * K;
  const unsigned short* arow1 = A + (size_t)(m0 + 16 + ln) * K;
  int aoff = laneHi ? 8 : 0;
  int boff = laneHi ? 8 : 0;
  v8f zero = {0.f, 0.f, 0.f, 0.f, 0.f, 0.f, 0.f, 0.f};
  v8f acc[2][4] = {{zero, zero, zero, zero}, {zero, zero, zero, zero}};

  const unsigned short* Wt = W + (size_t)n0 * K;
  stage_b_tile(Wt, K, Bs[0], tid);
  stage_b_wait(tid);
  __syncthreads();

  int nK = K >> 5;
  for (int i = 0; i < nK; ++i) {
    int cur = i & 1;
    if (i + 1 < nK) stage_b_tile(Wt + ((i + 1) << 5), K, Bs[cur ^ 1], tid);
    int kb = i << 5;
    v16bf a0 = load_frag(arow0 + kb + aoff, arow0 + kb + aoff + 16);
    v16bf a1 = load_frag(arow1 + kb + aoff, arow1 + kb + aoff + 16);
#pragma unroll
    for (int nt = 0; nt < 4; ++nt) {
      const unsigned short* brow = &Bs[cur][nt * 16 + ln][boff];
      v16bf bfv = load_frag(brow, brow + 16);
      acc[0][nt] = wmma_bf16(a0, bfv, acc[0][nt]);
      acc[1][nt] = wmma_bf16(a1, bfv, acc[1][nt]);
    }
    if (i + 1 < nK) stage_b_wait(tid);
    __syncthreads();
  }

#pragma unroll
  for (int mf = 0; mf < 2; ++mf) {
#pragma unroll
    for (int nt = 0; nt < 4; ++nt) {
      int n = n0 + nt * 16 + ln;
      float bv = bias[n];
#pragma unroll
      for (int r = 0; r < 8; ++r) {
        int m = m0 + mf * 16 + r + (laneHi << 3);
        float val = acc[mf][nt][r] + bv;
        if (MODE == 0) {
          outb[(size_t)m * N + n] = f2bf(val);
        } else if (MODE == 2) {
          val = 0.5f * val * (1.f + erff(val * 0.70710678118654752f));
          outb[(size_t)m * N + n] = f2bf(val);
        } else if (MODE == 1) {
          int w = m >> 6, t = m & 63;
          int b = w >> 4, wi = w & 15;
          int hp = ((wi >> 2) << 3) + (t >> 3);
          int wp = ((wi & 3) << 3) + (t & 7);
          int h = (hp + 4) & 31, ww = (wp + 4) & 31;  // roll(+4) back
          size_t xi = ((size_t)b * 1024 + h * 32 + ww) * 512 + n;
          float g = ada[(size_t)b * 3072 + 1024 + n];  // g_msa
          outf[xi] = xin[xi] + g * val;
        } else {  // MODE 3
          int b = m >> 10;
          size_t xi = (size_t)m * 512 + n;
          float g = ada[(size_t)b * 3072 + 2560 + n];  // g_mlp
          outf[xi] = xin[xi] + g * val;
        }
      }
    }
  }
}

// ---------------------------------------------------------------------------
// Windowed attention. 1 block = 1 (window, head); 4 waves, wave m-tile = 16.
// S = (q*scale) k^T + rpb + shift-mask ; softmax ; O = P v.
// ---------------------------------------------------------------------------
__device__ __forceinline__ int sgrp(int v) { return v < 24 ? 0 : (v < 28 ? 1 : 2); }
__device__ __forceinline__ int wlabel(int wi, int t) {
  int hp = ((wi >> 2) << 3) + (t >> 3);
  int wp = ((wi & 3) << 3) + (t & 7);
  return sgrp(hp) * 3 + sgrp(wp);
}

__global__ __launch_bounds__(128) void attn_kernel(
    const unsigned short* __restrict__ qkv, const float* __restrict__ rpb,
    unsigned short* __restrict__ attn_o) {
  const float scale = 0.17677669529663689f;  // 32^-0.5
  int wh = blockIdx.x;
  int win = wh >> 4, head = wh & 15;
  int wi = win & 15;
  int wave = threadIdx.x >> 5, lane = threadIdx.x & 31;
  int laneHi = lane >> 4, ln = lane & 15;

  __shared__ __align__(16) unsigned short P[64][64];
  __shared__ __align__(16) unsigned short Vs[64][32];

  const unsigned short* base = qkv + (size_t)win * 64 * 1536 + head * 32;

  // stage V[64 tok][32 d] into LDS (contiguous 16B chunks)
  for (int i = threadIdx.x; i < 256; i += 128) {
    int row = i >> 2, part = i & 3;
    *(U4*)&Vs[row][part * 8] =
        *(const U4*)(base + 1024 + (size_t)row * 1536 + part * 8);
  }
  __syncthreads();

  v8f zero = {0.f, 0.f, 0.f, 0.f, 0.f, 0.f, 0.f, 0.f};
  int aoff = laneHi ? 8 : 0;

  // S = q k^T  (K = hd = 32, single WMMA per tile)
  const unsigned short* qrow = base + (size_t)(wave * 16 + ln) * 1536;
  v16bf aq = load_frag(qrow + aoff, qrow + aoff + 16);
  v8f s[4];
#pragma unroll
  for (int nt = 0; nt < 4; ++nt) {
    const unsigned short* krow =
        base + 512 + (size_t)(nt * 16 + ln) * 1536 + (laneHi ? 8 : 0);
    v16bf bk = load_frag(krow, krow + 16);
    s[nt] = wmma_bf16(aq, bk, zero);
  }

  // scale + relative position bias + shift mask
#pragma unroll
  for (int nt = 0; nt < 4; ++nt) {
    int n = nt * 16 + ln;
    int nlab = wlabel(wi, n);
    int nh = n >> 3, nw_ = n & 7;
#pragma unroll
    for (int r = 0; r < 8; ++r) {
      int m = wave * 16 + r + (laneHi << 3);
      int idx = ((m >> 3) - nh + 7) * 15 + ((m & 7) - nw_ + 7);
      float v = s[nt][r] * scale + rpb[idx * 16 + head];
      if (wlabel(wi, m) != nlab) v -= 100.f;
      s[nt][r] = v;
    }
  }

  // row softmax: rows live in 16-lane halves; xor 1,2,4,8 stays within half
#pragma unroll
  for (int r = 0; r < 8; ++r) {
    float mx = fmaxf(fmaxf(s[0][r], s[1][r]), fmaxf(s[2][r], s[3][r]));
    mx = fmaxf(mx, __shfl_xor(mx, 1));
    mx = fmaxf(mx, __shfl_xor(mx, 2));
    mx = fmaxf(mx, __shfl_xor(mx, 4));
    mx = fmaxf(mx, __shfl_xor(mx, 8));
    float e[4], sum = 0.f;
#pragma unroll
    for (int nt = 0; nt < 4; ++nt) { e[nt] = __expf(s[nt][r] - mx); sum += e[nt]; }
    sum += __shfl_xor(sum, 1);
    sum += __shfl_xor(sum, 2);
    sum += __shfl_xor(sum, 4);
    sum += __shfl_xor(sum, 8);
    float inv = 1.f / sum;
    int m = wave * 16 + r + (laneHi << 3);
#pragma unroll
    for (int nt = 0; nt < 4; ++nt) P[m][nt * 16 + ln] = f2bf(e[nt] * inv);
  }
  // each wave re-reads only rows it wrote; compiler inserts DS waits

  // O = P V   (M=16/wave, N=32, K=64 -> 2 K-chunks)
  v8f o[2] = {zero, zero};
#pragma unroll
  for (int kc = 0; kc < 2; ++kc) {
    const unsigned short* prow = &P[wave * 16 + ln][kc * 32];
    v16bf pa = load_frag(prow + (laneHi ? 16 : 0), prow + (laneHi ? 16 : 0) + 8);
#pragma unroll
    for (int nv = 0; nv < 2; ++nv) {
      FragBF bv;
      int k0 = kc * 32 + (laneHi ? 16 : 0);
      int d = nv * 16 + ln;
#pragma unroll
      for (int e2 = 0; e2 < 16; ++e2) bv.s[e2] = Vs[k0 + e2][d];
      o[nv] = wmma_bf16(pa, bv.v, o[nv]);
    }
  }
#pragma unroll
  for (int nv = 0; nv < 2; ++nv) {
#pragma unroll
    for (int r = 0; r < 8; ++r) {
      int m = wave * 16 + r + (laneHi << 3);
      attn_o[((size_t)win * 64 + m) * 512 + head * 32 + nv * 16 + ln] =
          f2bf(o[nv][r]);
    }
  }
}

// ---------------------------------------------------------------------------
// Launch
// ---------------------------------------------------------------------------
extern "C" void kernel_launch(void* const* d_in, const int* in_sizes, int n_in,
                              void* d_out, int out_size, void* d_ws,
                              size_t ws_size, hipStream_t stream) {
  (void)in_sizes; (void)n_in; (void)out_size; (void)ws_size;
  const float* x       = (const float*)d_in[0];
  const float* cond    = (const float*)d_in[1];
  const float* norm1_w = (const float*)d_in[2];
  const float* norm1_b = (const float*)d_in[3];
  const float* qkv_w   = (const float*)d_in[4];
  const float* qkv_b   = (const float*)d_in[5];
  const float* rpb     = (const float*)d_in[6];
  const float* proj_w  = (const float*)d_in[7];
  const float* proj_b  = (const float*)d_in[8];
  const float* norm2_w = (const float*)d_in[9];
  const float* norm2_b = (const float*)d_in[10];
  const float* fc1_w   = (const float*)d_in[11];
  const float* fc1_b   = (const float*)d_in[12];
  const float* fc2_w   = (const float*)d_in[13];
  const float* fc2_b   = (const float*)d_in[14];
  const float* ada_w   = (const float*)d_in[15];
  const float* ada_b   = (const float*)d_in[16];

  char* p = (char*)d_ws;
  float* ada              = (float*)p;          p += 32ll * 3072 * 4;
  unsigned short* wq      = (unsigned short*)p; p += 1536ll * 512 * 2;
  unsigned short* wp      = (unsigned short*)p; p += 512ll * 512 * 2;
  unsigned short* w1      = (unsigned short*)p; p += 2048ll * 512 * 2;
  unsigned short* w2      = (unsigned short*)p; p += 512ll * 2048 * 2;
  unsigned short* hw      = (unsigned short*)p; p += 32768ll * 512 * 2;   // LN1 out (windowed)
  unsigned short* qkvb    = (unsigned short*)p; p += 32768ll * 1536 * 2;  // qkv out
  unsigned short* ao      = (unsigned short*)p; p += 32768ll * 512 * 2;   // attn out
  float* x2               = (float*)p;          p += 32768ll * 512 * 4;   // residual 1
  unsigned short* mlp_in  = hw;    // reuse: LN1 buffer dead after qkv GEMM
  unsigned short* h1      = qkvb;  // reuse: qkv+attn region (134 MB) for fc1 out

  ada_kernel<<<384, 256, 0, stream>>>(cond, ada_w, ada_b, ada);
  cvt_kernel<<<(1536 * 512) / 256, 256, 0, stream>>>(qkv_w, wq, 1536 * 512);
  cvt_kernel<<<(512 * 512) / 256, 256, 0, stream>>>(proj_w, wp, 512 * 512);
  cvt_kernel<<<(2048 * 512) / 256, 256, 0, stream>>>(fc1_w, w1, 2048 * 512);
  cvt_kernel<<<(512 * 2048) / 256, 256, 0, stream>>>(fc2_w, w2, 512 * 2048);

  prep_kernel<1><<<4096, 256, 0, stream>>>(x, ada, norm1_w, norm1_b, hw);
  gemm_bf16_kernel<0><<<dim3(24, 128), 256, 0, stream>>>(
      hw, wq, qkv_b, qkvb, nullptr, nullptr, nullptr, 32768, 1536, 512);
  attn_kernel<<<8192, 128, 0, stream>>>(qkvb, rpb, ao);
  gemm_bf16_kernel<1><<<dim3(8, 128), 256, 0, stream>>>(
      ao, wp, proj_b, nullptr, ada, x, x2, 32768, 512, 512);
  prep_kernel<2><<<4096, 256, 0, stream>>>(x2, ada, norm2_w, norm2_b, mlp_in);
  gemm_bf16_kernel<2><<<dim3(32, 128), 256, 0, stream>>>(
      mlp_in, w1, fc1_b, h1, nullptr, nullptr, nullptr, 32768, 2048, 512);
  gemm_bf16_kernel<3><<<dim3(8, 128), 256, 0, stream>>>(
      h1, w2, fc2_b, nullptr, ada, x2, (float*)d_out, 32768, 512, 2048);
}
